// SSL_2d_9371618639977
// MI455X (gfx1250) — matrix operand order repaired and checked by reading, not compiled, not run on hardware
//
#include <hip/hip_runtime.h>
#include <hip/hip_bf16.h>
#include <stdint.h>

// MI455X (gfx1250, wave32) implementation.
// Structure:
//   k1: W1 f32 -> f16 (one-time, 256KB -> 128KB in d_ws)
//   k2: s[b,n] = silu(h @ W1^T + b1) . w2   via v_wmma_f32_16x16x32_f16
//   k3: out[b,i,j] = (int64)(s[b,i] - s[b,j] + b2)
//       268MB write-once stream -> non-temporal b128 stores (don't thrash L2;
//       keep the 64KB, 2048x-reused s[] array resident instead).

typedef __attribute__((ext_vector_type(16))) _Float16 v16h;
typedef __attribute__((ext_vector_type(8)))  float    v8f;
typedef __attribute__((ext_vector_type(2)))  long long v2ll;

#define B_ 8
#define N_ 2048
#define H_ 256
#define ROWS (B_ * N_)   // 16384

// ---------------- Kernel 1: convert W1 to f16 ----------------
__global__ __launch_bounds__(256) void cvt_w1_f16(const float* __restrict__ W1,
                                                  _Float16* __restrict__ W1h) {
    int i = blockIdx.x * blockDim.x + threadIdx.x;
    if (i < H_ * H_) W1h[i] = (_Float16)W1[i];
}

// ---------------- Kernel 2: WMMA GEMM + SiLU + dot(w2) ----------------
// One wave handles 16 consecutive rows of h. A tile (16x256 f16) lives in
// registers (64 VGPRs/lane); 16 o-tiles x 8 k-chunks = 128 WMMAs per wave.
__global__ __launch_bounds__(256) void gemm_silu_dot(
    const float*    __restrict__ hptr,
    const _Float16* __restrict__ W1h,
    const float*    __restrict__ b1,
    const float*    __restrict__ w2,
    float*          __restrict__ sOut)
{
    const int lane = threadIdx.x & 31;
    const int wave = (blockIdx.x * blockDim.x + threadIdx.x) >> 5;
    const int row0 = wave * 16;          // 1024 waves * 16 rows = 16384
    const int kh   = lane >> 4;          // half-wave select
    const int mloc = lane & 15;          // A-row / B-column / D-column index

    // ---- Load A: 16x256 f16, contiguous k per lane half, same order as B ----
    // VGPRs 0..7 hold K = 16*kh + {0..15} (contiguous, matching B)
    v16h A[8];
    const float* hrow = hptr + (size_t)(row0 + mloc) * H_;
    #pragma unroll
    for (int kc = 0; kc < 8; ++kc) {
        const int ka = kc * 32 + 16 * kh;       // first run of 8
        const int kb = kc * 32 + 16 * kh + 8;   // second run of 8
        const float4 fa0 = *(const float4*)(hrow + ka);
        const float4 fa1 = *(const float4*)(hrow + ka + 4);
        const float4 fb0 = *(const float4*)(hrow + kb);
        const float4 fb1 = *(const float4*)(hrow + kb + 4);
        A[kc][0]  = (_Float16)fa0.x; A[kc][1]  = (_Float16)fa0.y;
        A[kc][2]  = (_Float16)fa0.z; A[kc][3]  = (_Float16)fa0.w;
        A[kc][4]  = (_Float16)fa1.x; A[kc][5]  = (_Float16)fa1.y;
        A[kc][6]  = (_Float16)fa1.z; A[kc][7]  = (_Float16)fa1.w;
        A[kc][8]  = (_Float16)fb0.x; A[kc][9]  = (_Float16)fb0.y;
        A[kc][10] = (_Float16)fb0.z; A[kc][11] = (_Float16)fb0.w;
        A[kc][12] = (_Float16)fb1.x; A[kc][13] = (_Float16)fb1.y;
        A[kc][14] = (_Float16)fb1.z; A[kc][15] = (_Float16)fb1.w;
    }

    float rowacc[8] = {0.f, 0.f, 0.f, 0.f, 0.f, 0.f, 0.f, 0.f};

    // ---- 16 output-channel tiles of 16 ----
    for (int ot = 0; ot < 16; ++ot) {
        const int c0 = ot * 16;
        v8f acc = {};
        // B layout: lane's column = mloc; K 0..15 (half 0) / 16..31 (half 1),
        // contiguous along a W1 row -> one 32-byte contiguous f16 load.
        const _Float16* wrow = W1h + (size_t)(c0 + mloc) * H_ + 16 * kh;
        #pragma unroll
        for (int kc = 0; kc < 8; ++kc) {
            const v16h Bv = *(const v16h*)(wrow + kc * 32);
            acc = __builtin_amdgcn_wmma_f32_16x16x32_f16(
                false, A[kc], false, Bv, (short)0, acc, false, false);
        }
        // Epilogue fused into accumulator: +b1, SiLU, *w2, accumulate per-row.
        const float bb = b1[c0 + mloc];
        const float ww = w2[c0 + mloc];
        #pragma unroll
        for (int i = 0; i < 8; ++i) {
            const float xv = acc[i] + bb;
            const float sg = 1.0f / (1.0f + __expf(-xv));
            rowacc[i] += xv * sg * ww;
        }
    }

    // ---- Cross-lane reduction over the 16 columns held per half-wave ----
    #pragma unroll
    for (int m = 1; m <= 8; m <<= 1) {
        #pragma unroll
        for (int i = 0; i < 8; ++i)
            rowacc[i] += __shfl_xor(rowacc[i], m, 32);
    }
    // D layout: lanes 0-15 VGPR i -> row row0+i; lanes 16-31 -> row row0+8+i.
    if (mloc == 0) {
        float* dst = sOut + row0 + 8 * kh;
        #pragma unroll
        for (int i = 0; i < 8; ++i) dst[i] = rowacc[i];
    }
}

// ---------------- Kernel 3: pairwise difference, int64 NT store stream ----------------
// One block per output row (b,i): 256 threads x 8 elements = 2048 columns.
__global__ __launch_bounds__(256) void pairwise_out(
    const float* __restrict__ s,
    const float* __restrict__ b2p,
    long long*   __restrict__ out)
{
    const float b2  = b2p[0];
    const int   row = blockIdx.x;               // b*N + i, 0..16383
    const int   b   = row >> 11;                // / 2048
    const int   j0  = (int)threadIdx.x * 8;

    const float  si  = s[row] + b2;             // uniform -> scalar load
    const float* srb = s + (size_t)b * N_;
    const float4 sj0 = *(const float4*)(srb + j0);
    const float4 sj1 = *(const float4*)(srb + j0 + 4);

    long long* o = out + (size_t)row * N_ + j0;
    v2ll r;
    // trunc-toward-zero == torch .long()
    r.x = (long long)(si - sj0.x); r.y = (long long)(si - sj0.y);
    __builtin_nontemporal_store(r, (v2ll*)(o));
    r.x = (long long)(si - sj0.z); r.y = (long long)(si - sj0.w);
    __builtin_nontemporal_store(r, (v2ll*)(o + 2));
    r.x = (long long)(si - sj1.x); r.y = (long long)(si - sj1.y);
    __builtin_nontemporal_store(r, (v2ll*)(o + 4));
    r.x = (long long)(si - sj1.z); r.y = (long long)(si - sj1.w);
    __builtin_nontemporal_store(r, (v2ll*)(o + 6));
}

extern "C" void kernel_launch(void* const* d_in, const int* in_sizes, int n_in,
                              void* d_out, int out_size, void* d_ws, size_t ws_size,
                              hipStream_t stream) {
    // setup_inputs order: h, node_mask, n_nodes, W1, b1, w2, b2
    const float* h  = (const float*)d_in[0];
    const float* W1 = (const float*)d_in[3];
    const float* b1 = (const float*)d_in[4];
    const float* w2 = (const float*)d_in[5];
    const float* b2 = (const float*)d_in[6];

    // Workspace: [0, 128KB) W1 in f16; [128KB, 128KB+64KB) s[B*N] f32.
    _Float16* W1h = (_Float16*)d_ws;
    float*    s   = (float*)((char*)d_ws + (size_t)H_ * H_ * sizeof(_Float16));
    long long* out = (long long*)d_out;

    cvt_w1_f16<<<(H_ * H_ + 255) / 256, 256, 0, stream>>>(W1, W1h);
    gemm_silu_dot<<<(ROWS / 16) / 8, 256, 0, stream>>>(h, W1h, b1, w2, s);
    pairwise_out<<<ROWS, 256, 0, stream>>>(s, b2, out);
}
